// min_model_85830626443599
// MI455X (gfx1250) — matrix operand order repaired and checked by reading, not compiled, not run on hardware
//
#include <hip/hip_runtime.h>
#include <hip/hip_bf16.h>

#define TL 2048  // sequence length
#define TV 2048  // vocab / channels
#define TB 4     // batch

typedef __attribute__((ext_vector_type(16))) __bf16 v16bf;
typedef __attribute__((ext_vector_type(8)))  float  v8f;

__device__ __forceinline__ unsigned short f2bf(float x) {
  union { float f; unsigned int u; } c; c.f = x;
  unsigned int r = c.u + 0x7FFFu + ((c.u >> 16) & 1u);   // RNE
  return (unsigned short)(r >> 16);
}

// ---------------------------------------------------------------------------
// C[M,N] = A[M,K](bf16,row-major) @ B[N,K](bf16,row-major)^T
// 256 threads = 8 waves; block tile 128x128; wave tile 32x64.
// Triple-buffered LDS fed by GLOBAL_LOAD_ASYNC_TO_LDS_B128 (ASYNCcnt),
// tiles prefetched two k-steps ahead; v_wmma_f32_16x16x32_bf16 inner loop.
// Optional per-output-row scale and bf16 output, batched via blockIdx.z.
// ---------------------------------------------------------------------------
#define LDS_ROW 40                      // 32 elems + pad (80B = 20 banks, conflict-free)
#define TILE_ELEMS (128 * LDS_ROW)      // 5120 bf16 per buffer
#define TILE_BYTES (TILE_ELEMS * 2)     // 10240 B

template <int OUT_BF16, int ROW_SCALE>
__global__ __launch_bounds__(256) void gemm_bt(
    const unsigned short* __restrict__ Ag, const unsigned short* __restrict__ Bg,
    void* __restrict__ Cg, const float* __restrict__ rsg, int M, int N, int K,
    long long sA, long long sB, long long sC, long long sS) {
  __shared__ alignas(16) unsigned short lA[3][TILE_ELEMS];
  __shared__ alignas(16) unsigned short lB[3][TILE_ELEMS];

  const unsigned short* A = Ag + (long long)blockIdx.z * sA;
  const unsigned short* B = Bg + (long long)blockIdx.z * sB;

  const int tid = threadIdx.x;
  const int wave = tid >> 5, lane = tid & 31;
  const int hf = lane >> 4, l16 = lane & 15;
  const int wm = wave & 3, wn = wave >> 2;  // 4 waves along M, 2 along N
  const int bm = blockIdx.y * 128, bn = blockIdx.x * 128;

  // Per-thread staging chunks: 512 chunks of 8 bf16 per 128x32 tile, 2 per thread.
  const int c0 = tid, c1 = tid + 256;
  const int r0 = c0 >> 2, o0 = (c0 & 3) << 3;
  const int r1 = c1 >> 2, o1 = (c1 & 3) << 3;
  // Global element offsets (kk added per tile), fit in 32 bits after *2.
  const long long gA0 = (long long)(bm + r0) * K + o0;
  const long long gA1 = (long long)(bm + r1) * K + o1;
  const long long gB0 = (long long)(bn + r0) * K + o0;
  const long long gB1 = (long long)(bn + r1) * K + o1;
  // LDS byte addresses of this thread's chunks (buffer 0).
  const unsigned dA0 = (unsigned)(unsigned long long)&lA[0][r0 * LDS_ROW + o0];
  const unsigned dA1 = (unsigned)(unsigned long long)&lA[0][r1 * LDS_ROW + o1];
  const unsigned dB0 = (unsigned)(unsigned long long)&lB[0][r0 * LDS_ROW + o0];
  const unsigned dB1 = (unsigned)(unsigned long long)&lB[0][r1 * LDS_ROW + o1];

#define ASYNC_B128(lds, voff, sbase)                                  \
  asm volatile("global_load_async_to_lds_b128 %0, %1, %2"             \
               :: "v"(lds), "v"(voff), "s"(sbase) : "memory")

  auto issueTile = [&](int buf, int kk) {
    const unsigned bo = (unsigned)buf * TILE_BYTES;
    ASYNC_B128(dA0 + bo, (unsigned)((gA0 + kk) << 1), A);
    ASYNC_B128(dA1 + bo, (unsigned)((gA1 + kk) << 1), A);
    ASYNC_B128(dB0 + bo, (unsigned)((gB0 + kk) << 1), B);
    ASYNC_B128(dB1 + bo, (unsigned)((gB1 + kk) << 1), B);
  };

  v8f acc[2][4];
#pragma unroll
  for (int im = 0; im < 2; ++im)
#pragma unroll
    for (int in = 0; in < 4; ++in)
      acc[im][in] = (v8f){0.f, 0.f, 0.f, 0.f, 0.f, 0.f, 0.f, 0.f};

  const int NK = K >> 5;
  issueTile(0, 0);
  if (NK > 1) issueTile(1, 32);

  int cur = 0, pre = 2;
  for (int kt = 0; kt < NK; ++kt) {
    // Retire tile kt (async loads complete in issue order); keep tile kt+1 in flight.
    if (kt + 1 < NK) asm volatile("s_wait_asynccnt 0x4" ::: "memory");
    else             asm volatile("s_wait_asynccnt 0x0" ::: "memory");
    __syncthreads();  // publish tile kt to all waves; prior reads of buffer `pre` done
    if (kt + 2 < NK) issueTile(pre, (kt + 2) << 5);

    union FR { uint4 q[2]; v16bf v; };
    FR af[2], bfr[4];
#pragma unroll
    for (int im = 0; im < 2; ++im) {  // A 16x32: lane half hf holds K [8h,8h+8)+[16+8h,..)
      const unsigned short* p = &lA[cur][(wm * 32 + im * 16 + l16) * LDS_ROW + hf * 8];
      af[im].q[0] = *(const uint4*)p;
      af[im].q[1] = *(const uint4*)(p + 16);
    }
#pragma unroll
    for (int in = 0; in < 4; ++in) {  // B 32x16 fed K-major: lane half hf holds K [8h,8h+8)+[16+8h,..)
      const unsigned short* p = &lB[cur][(wn * 64 + in * 16 + l16) * LDS_ROW + hf * 8];
      bfr[in].q[0] = *(const uint4*)p;
      bfr[in].q[1] = *(const uint4*)(p + 16);
    }
#pragma unroll
    for (int im = 0; im < 2; ++im)
#pragma unroll
      for (int in = 0; in < 4; ++in)
        acc[im][in] = __builtin_amdgcn_wmma_f32_16x16x32_bf16(
            false, af[im].v, false, bfr[in].v, (short)0, acc[im][in], false, false);

    cur = (cur == 2) ? 0 : cur + 1;
    pre = (pre == 2) ? 0 : pre + 1;
  }

  // Epilogue. C/D layout: VGPR r, lane l -> M = r + 8*(l>>4), N = l&15.
  const int gm0 = bm + wm * 32, gn0 = bn + wn * 64;
#pragma unroll
  for (int im = 0; im < 2; ++im)
#pragma unroll
    for (int in = 0; in < 4; ++in)
#pragma unroll
      for (int r = 0; r < 8; ++r) {
        int m = gm0 + im * 16 + hf * 8 + r;
        int n = gn0 + in * 16 + l16;
        float v = acc[im][in][r];
        if constexpr (ROW_SCALE) v *= rsg[(long long)blockIdx.z * sS + m];
        long long o = (long long)blockIdx.z * sC + (long long)m * N + n;
        if constexpr (OUT_BF16)
          ((unsigned short*)Cg)[o] = f2bf(v);
        else
          ((float*)Cg)[o] = v;
      }
}

// --------------------------- small helper kernels ---------------------------

// out[i,j] = bf16(in[j,i]) for [TV,TV]
__global__ void k_cvt_t(const float* __restrict__ in, unsigned short* __restrict__ out) {
  __shared__ float t[32][33];
  int i0 = blockIdx.x * 32, j0 = blockIdx.y * 32, x = threadIdx.x;
  for (int r = threadIdx.y; r < 32; r += 8)
    t[r][x] = in[(long long)(j0 + r) * TV + i0 + x];
  __syncthreads();
  for (int r = threadIdx.y; r < 32; r += 8)
    out[(long long)(i0 + r) * TV + j0 + x] = f2bf(t[x][r]);
}

__global__ void k_cvt(const float* __restrict__ in, unsigned short* __restrict__ out, int n) {
  int i = blockIdx.x * 256 + threadIdx.x;
  if (i < n) out[i] = f2bf(in[i]);
}

// cs[v] = sum_j W[j,v]
__global__ void k_colsum(const float* __restrict__ W, float* __restrict__ cs) {
  int v = blockIdx.x * 256 + threadIdx.x;
  float s = 0.f;
  for (int j = 0; j < TV; ++j) s += W[(long long)j * TV + v];
  cs[v] = s;
}

// S0[b,q,k] = (k<=q) ? G0[iq,ik] + cs0[iq]*v0[q-k] : 0   (bf16)
__global__ void k_s0(const float* __restrict__ G0, const float* __restrict__ cs0,
                     const float* __restrict__ v0, const int* __restrict__ idx,
                     unsigned short* __restrict__ S0) {
  int k = blockIdx.x * 256 + threadIdx.x, q = blockIdx.y, b = blockIdx.z;
  long long o = ((long long)b * TL + q) * TL + k;
  if (k <= q) {
    int iq = idx[b * TL + q], ik = idx[b * TL + k];
    S0[o] = f2bf(G0[(long long)iq * TV + ik] + cs0[iq] * v0[q - k]);
  } else
    S0[o] = 0;
}

// out[b][j,k] = Wb[j, idx[b,k]]   (gathered V-operand, [TV x TL] bf16 per batch)
__global__ void k_gather(const unsigned short* __restrict__ Wb, const int* __restrict__ idx,
                         unsigned short* __restrict__ out) {
  int k = blockIdx.x * 256 + threadIdx.x, j = blockIdx.y, b = blockIdx.z;
  out[((long long)b * TV + j) * TL + k] = Wb[(long long)j * TV + idx[b * TL + k]];
}

__global__ void k_rs(const float* __restrict__ cs, const int* __restrict__ idx,
                     float* __restrict__ rs, int n) {
  int i = blockIdx.x * 256 + threadIdx.x;
  if (i < n) rs[i] = cs[idx[i]];
}

// P1m[q,k] = (k<=q) ? bf16(v1[q-k]) : 0
__global__ void k_p1(const float* __restrict__ v1, unsigned short* __restrict__ P) {
  int k = blockIdx.x * 256 + threadIdx.x, q = blockIdx.y;
  P[(long long)q * TL + k] = (k <= q) ? f2bf(v1[q - k]) : (unsigned short)0;
}

// batched bf16 transpose: in [TL,TV] -> out [TV,TL]
__global__ void k_tr_bf(const unsigned short* __restrict__ in, unsigned short* __restrict__ out) {
  __shared__ unsigned short t[32][33];
  long long base = (long long)blockIdx.z * TL * TV;
  int i0 = blockIdx.x * 32, j0 = blockIdx.y * 32, x = threadIdx.x;
  for (int r = threadIdx.y; r < 32; r += 8)
    t[r][x] = in[base + (long long)(j0 + r) * TV + i0 + x];
  __syncthreads();
  for (int r = threadIdx.y; r < 32; r += 8)
    out[base + (long long)(i0 + r) * TL + j0 + x] = t[x][r];
}

// A2[b,q,k] = (k<=q) ? t2[b,k,idx[b,q]] : 0
__global__ void k_a2(const unsigned short* __restrict__ t2, const int* __restrict__ idx,
                     unsigned short* __restrict__ A2) {
  int k = blockIdx.x * 256 + threadIdx.x, q = blockIdx.y, b = blockIdx.z;
  long long o = ((long long)b * TL + q) * TL + k;
  if (k <= q)
    A2[o] = t2[((long long)b * TL + k) * TV + idx[b * TL + q]];
  else
    A2[o] = 0;
}

// ---------------------------------------------------------------------------

extern "C" void kernel_launch(void* const* d_in, const int* in_sizes, int n_in,
                              void* d_out, int out_size, void* d_ws, size_t ws_size,
                              hipStream_t stream) {
  const int*   idx = (const int*)d_in[0];
  const float* Wq0 = (const float*)d_in[1];
  const float* Wk0 = (const float*)d_in[2];
  const float* Wv0 = (const float*)d_in[3];
  const float* Wk1 = (const float*)d_in[4];
  const float* Wv1 = (const float*)d_in[5];
  const float* Wq2 = (const float*)d_in[6];
  const float* Wk2 = (const float*)d_in[7];
  const float* Wv2 = (const float*)d_in[8];
  const float* v0  = (const float*)d_in[9];
  const float* v1  = (const float*)d_in[10];

  char* ws = (char*)d_ws;
  size_t off = 0;
  auto alloc = [&](size_t bytes) {
    off = (off + 255) & ~(size_t)255;
    void* p = ws + off;
    off += bytes;
    return p;
  };
  const size_t SZ_BLL = (size_t)TB * TL * TL * 2;
  const size_t SZ_VV2 = (size_t)TV * TV * 2;

  unsigned short* bufS = (unsigned short*)alloc(SZ_BLL);      // S0 -> A2
  unsigned short* bufG = (unsigned short*)alloc(SZ_BLL);      // Gv0 -> Gv2
  unsigned short* bufX = (unsigned short*)alloc(SZ_BLL);      // l0 -> V1t -> t2
  unsigned short* bufY = (unsigned short*)alloc(SZ_BLL);      // V1 -> l1
  unsigned short* T1   = (unsigned short*)alloc(SZ_VV2);      // Tq0 -> Tq2 -> P1m
  unsigned short* T2   = (unsigned short*)alloc(SZ_VV2);      // Tk0 -> Tk2
  float*          G0   = (float*)alloc((size_t)TV * TV * 4);
  unsigned short* M2b  = (unsigned short*)alloc(SZ_VV2);
  unsigned short* Wv0b = (unsigned short*)alloc(SZ_VV2);
  unsigned short* Wv1b = (unsigned short*)alloc(SZ_VV2);
  unsigned short* Wv2b = (unsigned short*)alloc(SZ_VV2);
  float* cs0 = (float*)alloc(TV * 4);
  float* cs1 = (float*)alloc(TV * 4);
  float* rs1 = (float*)alloc((size_t)TB * TL * 4);

  dim3 bt(32, 8), gt(TV / 32, TV / 32);
  dim3 gg1(TV / 128, TV / 128, 1);
  dim3 ggB(TV / 128, TL / 128, TB);
  const long long LL = (long long)TL * TL, LV = (long long)TL * TV, VL = (long long)TV * TL;

  // Gram matrices: G0 = Wq0^T @ Wk0 (f32), M2 = Wq2^T @ Wk2 (bf16)
  k_cvt_t<<<gt, bt, 0, stream>>>(Wq0, T1);
  k_cvt_t<<<gt, bt, 0, stream>>>(Wk0, T2);
  gemm_bt<0, 0><<<gg1, 256, 0, stream>>>(T1, T2, G0, nullptr, TV, TV, TV, 0, 0, 0, 0);
  k_cvt_t<<<gt, bt, 0, stream>>>(Wq2, T1);
  k_cvt_t<<<gt, bt, 0, stream>>>(Wk2, T2);
  gemm_bt<1, 0><<<gg1, 256, 0, stream>>>(T1, T2, M2b, nullptr, TV, TV, TV, 0, 0, 0, 0);

  // bf16 weight copies, column sums, per-row scales
  k_cvt<<<(TV * TV) / 256, 256, 0, stream>>>(Wv0, Wv0b, TV * TV);
  k_cvt<<<(TV * TV) / 256, 256, 0, stream>>>(Wv1, Wv1b, TV * TV);
  k_cvt<<<(TV * TV) / 256, 256, 0, stream>>>(Wv2, Wv2b, TV * TV);
  k_colsum<<<TV / 256, 256, 0, stream>>>(Wk0, cs0);
  k_colsum<<<TV / 256, 256, 0, stream>>>(Wk1, cs1);
  k_rs<<<(TB * TL) / 256, 256, 0, stream>>>(cs1, idx, rs1, TB * TL);

  // layer 0: S0 gather-build, l0 = S0 @ Gv0^T
  k_s0<<<dim3(TL / 256, TL, TB), 256, 0, stream>>>(G0, cs0, v0, idx, bufS);
  k_gather<<<dim3(TL / 256, TV, TB), 256, 0, stream>>>(Wv0b, idx, bufG);
  gemm_bt<1, 0><<<ggB, 256, 0, stream>>>(bufS, bufG, bufX, nullptr, TL, TV, TL, LL, VL, LV, 0);

  // layer 1: V1 = l0 @ Wv1^T; l1 = diag(rs1) * (trilP1 @ V1)
  gemm_bt<1, 0><<<ggB, 256, 0, stream>>>(bufX, Wv1b, bufY, nullptr, TL, TV, TV, LV, 0, LV, 0);
  k_tr_bf<<<dim3(TV / 32, TL / 32, TB), bt, 0, stream>>>(bufY, bufX);  // V1t
  k_p1<<<dim3(TL / 256, TL), 256, 0, stream>>>(v1, T1);                // trilP1
  gemm_bt<1, 1><<<ggB, 256, 0, stream>>>(T1, bufX, bufY, rs1, TL, TV, TL, 0, VL, LV, TL);

  // layer 2: t2 = l1 @ M2^T; A2 gather; logits = A2 @ Gv2^T
  gemm_bt<1, 0><<<ggB, 256, 0, stream>>>(bufY, M2b, bufX, nullptr, TL, TV, TV, LV, 0, LV, 0);
  k_a2<<<dim3(TL / 256, TL, TB), 256, 0, stream>>>(bufX, idx, bufS);
  k_gather<<<dim3(TL / 256, TV, TB), 256, 0, stream>>>(Wv2b, idx, bufG);
  gemm_bt<0, 0><<<ggB, 256, 0, stream>>>(bufS, bufG, (float*)d_out, nullptr, TL, TV, TL, LL, VL,
                                         LV, 0);
}
